// SwinTransformerBlock_48344151884393
// MI455X (gfx1250) — matrix operand order repaired and checked by reading, not compiled, not run on hardware
//
#include <hip/hip_runtime.h>
#include <stdint.h>

#define TOK  100352     // B*H*W tokens
#define CDIM 256
#define QKVN 768
#define FFN  1024
#define NWIN 2048
#define WA   49

typedef __attribute__((ext_vector_type(16))) __bf16 bf16x16;
typedef __attribute__((ext_vector_type(8)))  float  f32x8;

struct FragBytes { uint4 lo, hi; };

static __device__ __forceinline__ bf16x16 make_frag(uint4 lo, uint4 hi) {
    FragBytes f; f.lo = lo; f.hi = hi;
    return __builtin_bit_cast(bf16x16, f);
}
static __device__ __forceinline__ unsigned short f2bf(float x) {
    unsigned u = __float_as_uint(x);
    u += 0x7FFFu + ((u >> 16) & 1u);           // round-to-nearest-even
    return (unsigned short)(u >> 16);
}
static __device__ __forceinline__ float bf2f(unsigned h) {
    return __uint_as_float(h << 16);
}
static __device__ __forceinline__ float wave_sum32(float v) {
#pragma unroll
    for (int m = 16; m; m >>= 1) v += __shfl_xor(v, m, 32);
    return v;
}

// ---------------------------------------------------------------- weight cvt
__global__ void k_cvt_bf16(const float* __restrict__ in,
                           unsigned short* __restrict__ out, int n) {
    int i = blockIdx.x * blockDim.x + threadIdx.x;
    if (i < n) out[i] = f2bf(in[i]);
}

// ------------------------------------------- combined rel-bias + shift mask
__global__ void k_bias_mask(const float* __restrict__ mask,
                            const float* __restrict__ rpb,
                            const int* __restrict__ relidx,
                            float* __restrict__ comb) {
    const int TOT = 64 * 8 * WA * WA;
    int i = blockIdx.x * blockDim.x + threadIdx.x;
    if (i >= TOT) return;
    int qk = i % (WA * WA);
    int t  = i / (WA * WA);
    int h  = t & 7;
    int wm = t >> 3;
    comb[i] = mask[wm * WA * WA + qk] + rpb[relidx[qk] * 8 + h];
}

// --------------------------- LayerNorm over 256 ch, optional shift+partition
__global__ __launch_bounds__(256) void k_ln256(
    const float* __restrict__ in, const float* __restrict__ g,
    const float* __restrict__ bt, unsigned short* __restrict__ out, int permute)
{
    int lane = threadIdx.x & 31;
    int wave = threadIdx.x >> 5;
    long row = (long)blockIdx.x * 8 + wave;         // output row (window order if permute)
    long src = row;
    if (permute) {
        int w  = (int)(row / WA);
        int i  = (int)(row % WA);
        int b  = w >> 6;
        int wi = w & 63;
        int hs = (wi >> 3) * 7 + i / 7;
        int ws = (wi & 7) * 7 + i % 7;
        int hh = hs + 3; if (hh >= 56) hh -= 56;    // roll(-3): src = (dst+3)%56
        int w2 = ws + 3; if (w2 >= 56) w2 -= 56;
        src = ((long)b * 56 + hh) * 56 + w2;
    }
    const float4* p = (const float4*)(in + src * CDIM);
    float4 a0 = p[lane];
    float4 a1 = p[lane + 32];
    float s = a0.x + a0.y + a0.z + a0.w + a1.x + a1.y + a1.z + a1.w;
    float mean = wave_sum32(s) * (1.f / 256.f);
    float v = 0.f, d;
    d = a0.x - mean; v += d * d;  d = a0.y - mean; v += d * d;
    d = a0.z - mean; v += d * d;  d = a0.w - mean; v += d * d;
    d = a1.x - mean; v += d * d;  d = a1.y - mean; v += d * d;
    d = a1.z - mean; v += d * d;  d = a1.w - mean; v += d * d;
    float inv = rsqrtf(wave_sum32(v) * (1.f / 256.f) + 1e-5f);
    float4 g0 = ((const float4*)g)[lane];
    float4 g1 = ((const float4*)g)[lane + 32];
    float4 b0 = ((const float4*)bt)[lane];
    float4 b1 = ((const float4*)bt)[lane + 32];
    float y0 = (a0.x - mean) * inv * g0.x + b0.x;
    float y1 = (a0.y - mean) * inv * g0.y + b0.y;
    float y2 = (a0.z - mean) * inv * g0.z + b0.z;
    float y3 = (a0.w - mean) * inv * g0.w + b0.w;
    float y4 = (a1.x - mean) * inv * g1.x + b1.x;
    float y5 = (a1.y - mean) * inv * g1.y + b1.y;
    float y6 = (a1.z - mean) * inv * g1.z + b1.z;
    float y7 = (a1.w - mean) * inv * g1.w + b1.w;
    uint2 o0, o1;
    o0.x = (unsigned)f2bf(y0) | ((unsigned)f2bf(y1) << 16);
    o0.y = (unsigned)f2bf(y2) | ((unsigned)f2bf(y3) << 16);
    o1.x = (unsigned)f2bf(y4) | ((unsigned)f2bf(y5) << 16);
    o1.y = (unsigned)f2bf(y6) | ((unsigned)f2bf(y7) << 16);
    ((uint2*)(out + row * CDIM))[lane]      = o0;
    ((uint2*)(out + row * CDIM))[lane + 32] = o1;
}

// -------------------------------- LayerNorm over 1024 ch, bf16 in-place
__global__ __launch_bounds__(256) void k_ln1024(
    unsigned short* __restrict__ buf, const float* __restrict__ g,
    const float* __restrict__ bt)
{
    int lane = threadIdx.x & 31;
    int wave = threadIdx.x >> 5;
    long row = (long)blockIdx.x * 8 + wave;
    uint4* p = (uint4*)(buf + row * FFN);
    float vals[32];
#pragma unroll
    for (int j = 0; j < 4; ++j) {
        uint4 c = p[lane + j * 32];
        unsigned wd[4] = {c.x, c.y, c.z, c.w};
#pragma unroll
        for (int k = 0; k < 4; ++k) {
            vals[j * 8 + 2 * k]     = bf2f(wd[k] & 0xFFFFu);
            vals[j * 8 + 2 * k + 1] = bf2f(wd[k] >> 16);
        }
    }
    float s = 0.f;
#pragma unroll
    for (int e = 0; e < 32; ++e) s += vals[e];
    float mean = wave_sum32(s) * (1.f / 1024.f);
    float v = 0.f;
#pragma unroll
    for (int e = 0; e < 32; ++e) { float d = vals[e] - mean; v += d * d; }
    float inv = rsqrtf(wave_sum32(v) * (1.f / 1024.f) + 1e-5f);
#pragma unroll
    for (int j = 0; j < 4; ++j) {
        int ch0 = (lane + j * 32) * 8;
        unsigned wd[4];
#pragma unroll
        for (int k = 0; k < 4; ++k) {
            float y0 = (vals[j * 8 + 2 * k]     - mean) * inv * g[ch0 + 2 * k]     + bt[ch0 + 2 * k];
            float y1 = (vals[j * 8 + 2 * k + 1] - mean) * inv * g[ch0 + 2 * k + 1] + bt[ch0 + 2 * k + 1];
            wd[k] = (unsigned)f2bf(y0) | ((unsigned)f2bf(y1) << 16);
        }
        uint4 o; o.x = wd[0]; o.y = wd[1]; o.z = wd[2]; o.w = wd[3];
        p[lane + j * 32] = o;
    }
}

// ------------------------------------------------------------- bf16 GEMM
// C(M,N) = A(M,K) @ W(N,K)^T + bias, with fused epilogues.
// Workgroup tile 128x128, 4 waves, each wave a 64x64 tile (16 WMMAs/K-step).
enum { EP_BF16 = 0, EP_PROJ = 1, EP_GELU = 2, EP_FC2 = 3 };

template <int MODE, int K>
__global__ __launch_bounds__(128) void k_gemm(
    const unsigned short* __restrict__ A, const unsigned short* __restrict__ W,
    const float* __restrict__ bias, int N,
    unsigned short* __restrict__ outb, float* __restrict__ outf,
    const float* __restrict__ res)
{
    const int lane = threadIdx.x & 31;
    const int wave = threadIdx.x >> 5;         // 0..3 -> 2x2 wave grid
    const int l15 = lane & 15, lh = lane >> 4;
    const long row0 = (long)blockIdx.y * 128 + (wave >> 1) * 64;
    const int  col0 = blockIdx.x * 128 + (wave & 1) * 64;

    f32x8 acc[4][4] = {};
    const unsigned short* Ab = A + (row0 + l15) * (long)K + lh * 16;  // A frag: halves hold K +0..15 / +16..31
    const unsigned short* Wb = W + (long)(col0 + l15) * K + lh * 16;  // B frag: halves hold K +0..15 / +16..31

#pragma unroll 2
    for (int k0 = 0; k0 < K; k0 += 32) {
        if (k0 + 32 < K) {                      // gfx1250 global_prefetch_b8
            __builtin_prefetch(Ab + k0 + 32, 0, 3);
            __builtin_prefetch(Wb + k0 + 32, 0, 3);
        }
        bf16x16 af[4], bw[4];
#pragma unroll
        for (int t = 0; t < 4; ++t) {
            const unsigned short* pa = Ab + (long)t * 16 * K + k0;
            af[t] = make_frag(*(const uint4*)pa, *(const uint4*)(pa + 8));
            const unsigned short* pb = Wb + (long)t * 16 * K + k0;
            bw[t] = make_frag(*(const uint4*)pb, *(const uint4*)(pb + 8));
        }
#pragma unroll
        for (int ti = 0; ti < 4; ++ti)
#pragma unroll
            for (int tj = 0; tj < 4; ++tj)
                acc[ti][tj] = __builtin_amdgcn_wmma_f32_16x16x32_bf16(
                    false, af[ti], false, bw[tj], (short)0, acc[ti][tj], false, false);
    }

    float bv[4];
#pragma unroll
    for (int tj = 0; tj < 4; ++tj) bv[tj] = bias[col0 + tj * 16 + l15];

#pragma unroll
    for (int ti = 0; ti < 4; ++ti) {
#pragma unroll
        for (int r = 0; r < 8; ++r) {
            long m = row0 + ti * 16 + r + lh * 8;
            long tgt = m;
            if (MODE == EP_PROJ) {                 // window reverse + roll(+3,+3)
                int wq = (int)(m / WA), i = (int)(m % WA);
                int b = wq >> 6, wi = wq & 63;
                int hs = (wi >> 3) * 7 + i / 7;
                int ws = (wi & 7) * 7 + i % 7;
                int hh = hs + 3; if (hh >= 56) hh -= 56;
                int w2 = ws + 3; if (w2 >= 56) w2 -= 56;
                tgt = ((long)b * 56 + hh) * 56 + w2;
            }
#pragma unroll
            for (int tj = 0; tj < 4; ++tj) {
                int col = col0 + tj * 16 + l15;
                float v = acc[ti][tj][r] + bv[tj];
                if (MODE == EP_BF16) {
                    outb[m * N + col] = f2bf(v);
                } else if (MODE == EP_GELU) {
                    float gg = 0.5f * v * (1.f + erff(v * 0.70710678f));
                    outb[m * N + col] = f2bf(gg);
                } else if (MODE == EP_FC2) {
                    long o = m * N + col;
                    outf[o] = v + res[o];
                } else {  // EP_PROJ
                    long o = tgt * CDIM + col;
                    outf[o] = v + res[o];
                }
            }
        }
    }
}

// ---------------------------------------------------- attention, 1 wave/(w,h)
__global__ __launch_bounds__(32) void k_attn(
    const unsigned short* __restrict__ qkv, const float* __restrict__ comb,
    unsigned short* __restrict__ outb)
{
    __shared__ __align__(16) unsigned short sP[64 * 64];   // softmax probs (bf16)
    __shared__ __align__(16) unsigned short sVT[32 * 72];  // V^T, pitch 72 (16B-aligned rows)
    const int lane = threadIdx.x;
    const int l15 = lane & 15, lh = lane >> 4;
    const int w = blockIdx.x >> 3;
    const int h = blockIdx.x & 7;
    const unsigned short* base = qkv + (size_t)w * WA * QKVN;

    // stage V^T: coalesced row reads, scattered b16 LDS stores
#pragma unroll
    for (int it = 0; it < 2; ++it) {
        int rv = it * 32 + lane;
        const unsigned short* pv = base + (size_t)rv * QKVN + 512 + h * 32;
        uint4 c0 = *(const uint4*)pv;
        uint4 c1 = *(const uint4*)(pv + 8);
        uint4 c2 = *(const uint4*)(pv + 16);
        uint4 c3 = *(const uint4*)(pv + 24);
        unsigned wd[16] = {c0.x, c0.y, c0.z, c0.w, c1.x, c1.y, c1.z, c1.w,
                           c2.x, c2.y, c2.z, c2.w, c3.x, c3.y, c3.z, c3.w};
#pragma unroll
        for (int dd = 0; dd < 32; ++dd) {
            unsigned u = wd[dd >> 1];
            sVT[dd * 72 + rv] = (unsigned short)((dd & 1) ? (u >> 16) : (u & 0xFFFFu));
        }
    }

    // S = q @ k^T  (K = HEAD_DIM = 32, one WMMA K-step)
    bf16x16 aq[4], bk[4];
#pragma unroll
    for (int t = 0; t < 4; ++t) {
        const unsigned short* pa = base + (size_t)(t * 16 + l15) * QKVN + h * 32 + lh * 16;
        aq[t] = make_frag(*(const uint4*)pa, *(const uint4*)(pa + 8));
        const unsigned short* pb = base + (size_t)(t * 16 + l15) * QKVN + 256 + h * 32 + lh * 16;
        bk[t] = make_frag(*(const uint4*)pb, *(const uint4*)(pb + 8));
    }
    f32x8 s[4][4] = {};
#pragma unroll
    for (int ti = 0; ti < 4; ++ti)
#pragma unroll
        for (int tj = 0; tj < 4; ++tj)
            s[ti][tj] = __builtin_amdgcn_wmma_f32_16x16x32_bf16(
                false, aq[ti], false, bk[tj], (short)0, s[ti][tj], false, false);

    // scale + bias + mask + row softmax (rows live in 16-lane half-groups)
    const float* cb = comb + ((size_t)(w & 63) * 8 + h) * (WA * WA);
#pragma unroll
    for (int ti = 0; ti < 4; ++ti) {
#pragma unroll
        for (int r = 0; r < 8; ++r) {
            int row = ti * 16 + r + lh * 8;
            float e[4];
            float mx = -3.0e38f;
#pragma unroll
            for (int tj = 0; tj < 4; ++tj) {
                int col = tj * 16 + l15;
                float sv = s[ti][tj][r] * 0.17677669529663689f;  // 32^-0.5
                sv = (row < WA && col < WA) ? (sv + cb[row * WA + col]) : -30000.f;
                e[tj] = sv;
                mx = fmaxf(mx, sv);
            }
            mx = fmaxf(mx, __shfl_xor(mx, 1, 32));
            mx = fmaxf(mx, __shfl_xor(mx, 2, 32));
            mx = fmaxf(mx, __shfl_xor(mx, 4, 32));
            mx = fmaxf(mx, __shfl_xor(mx, 8, 32));
            float sum = 0.f;
#pragma unroll
            for (int tj = 0; tj < 4; ++tj) { e[tj] = __expf(e[tj] - mx); sum += e[tj]; }
            sum += __shfl_xor(sum, 1, 32);
            sum += __shfl_xor(sum, 2, 32);
            sum += __shfl_xor(sum, 4, 32);
            sum += __shfl_xor(sum, 8, 32);
            float invs = 1.0f / sum;
#pragma unroll
            for (int tj = 0; tj < 4; ++tj)
                sP[row * 64 + tj * 16 + l15] = f2bf(e[tj] * invs);
        }
    }
    __syncthreads();

    // O = P @ V  (K = 64 keys, two WMMA K-steps)
    f32x8 o[4][2] = {};
    for (int k0 = 0; k0 < 64; k0 += 32) {
        bf16x16 ap[4], bv[2];
#pragma unroll
        for (int t = 0; t < 4; ++t) {
            const unsigned short* pa = sP + (t * 16 + l15) * 64 + k0 + lh * 16;
            ap[t] = make_frag(*(const uint4*)pa, *(const uint4*)(pa + 8));
        }
#pragma unroll
        for (int t = 0; t < 2; ++t) {
            const unsigned short* pb = sVT + (t * 16 + l15) * 72 + k0 + lh * 16;
            bv[t] = make_frag(*(const uint4*)pb, *(const uint4*)(pb + 8));
        }
#pragma unroll
        for (int ti = 0; ti < 4; ++ti)
#pragma unroll
            for (int tj = 0; tj < 2; ++tj)
                o[ti][tj] = __builtin_amdgcn_wmma_f32_16x16x32_bf16(
                    false, ap[ti], false, bv[tj], (short)0, o[ti][tj], false, false);
    }

#pragma unroll
    for (int ti = 0; ti < 4; ++ti) {
#pragma unroll
        for (int r = 0; r < 8; ++r) {
            int row = ti * 16 + r + lh * 8;
            if (row < WA) {
#pragma unroll
                for (int tj = 0; tj < 2; ++tj)
                    outb[((size_t)w * WA + row) * CDIM + h * 32 + tj * 16 + l15] =
                        f2bf(o[ti][tj][r]);
            }
        }
    }
}

// -------------------------------------------------------------------- launch
extern "C" void kernel_launch(void* const* d_in, const int* in_sizes, int n_in,
                              void* d_out, int out_size, void* d_ws, size_t ws_size,
                              hipStream_t stream)
{
    (void)in_sizes; (void)n_in; (void)out_size; (void)ws_size;
    const float* x      = (const float*)d_in[0];
    const float* ln1_w  = (const float*)d_in[1];
    const float* ln1_b  = (const float*)d_in[2];
    const float* qkv_w  = (const float*)d_in[3];
    const float* qkv_b  = (const float*)d_in[4];
    const float* rpb    = (const float*)d_in[5];
    const float* proj_w = (const float*)d_in[6];
    const float* proj_b = (const float*)d_in[7];
    const float* ln2_w  = (const float*)d_in[8];
    const float* ln2_b  = (const float*)d_in[9];
    const float* fc1_w  = (const float*)d_in[10];
    const float* fc1_b  = (const float*)d_in[11];
    const float* ln3_w  = (const float*)d_in[12];
    const float* ln3_b  = (const float*)d_in[13];
    const float* fc2_w  = (const float*)d_in[14];
    const float* fc2_b  = (const float*)d_in[15];
    const float* amask  = (const float*)d_in[16];
    const int*   relidx = (const int*)d_in[17];

    uint8_t* ws = (uint8_t*)d_ws;
    size_t off = 0;
    auto carve = [&](size_t bytes) -> uint8_t* {
        uint8_t* p = ws + off;
        off += (bytes + 255) & ~(size_t)255;
        return p;
    };
    unsigned short* bufA   = (unsigned short*)carve((size_t)TOK * CDIM * 2);  // winA, reused as z
    unsigned short* bufO   = (unsigned short*)carve((size_t)TOK * CDIM * 2);  // attn out
    float*          x1     = (float*)carve((size_t)TOK * CDIM * 4);
    float*          comb   = (float*)carve((size_t)64 * 8 * WA * WA * 4);
    unsigned short* wq     = (unsigned short*)carve((size_t)QKVN * CDIM * 2);
    unsigned short* wp     = (unsigned short*)carve((size_t)CDIM * CDIM * 2);
    unsigned short* wf1    = (unsigned short*)carve((size_t)FFN * CDIM * 2);
    unsigned short* wf2    = (unsigned short*)carve((size_t)CDIM * FFN * 2);
    unsigned short* bufQKV = (unsigned short*)carve((size_t)TOK * FFN * 2);   // qkv, reused as h

    // weights -> bf16, combined bias table
    k_cvt_bf16<<<(QKVN * CDIM + 255) / 256, 256, 0, stream>>>(qkv_w, wq, QKVN * CDIM);
    k_cvt_bf16<<<(CDIM * CDIM + 255) / 256, 256, 0, stream>>>(proj_w, wp, CDIM * CDIM);
    k_cvt_bf16<<<(FFN * CDIM + 255) / 256, 256, 0, stream>>>(fc1_w, wf1, FFN * CDIM);
    k_cvt_bf16<<<(CDIM * FFN + 255) / 256, 256, 0, stream>>>(fc2_w, wf2, CDIM * FFN);
    k_bias_mask<<<(64 * 8 * WA * WA + 255) / 256, 256, 0, stream>>>(amask, rpb, relidx, comb);

    // LN1 + roll(-3,-3) + window partition -> bf16
    k_ln256<<<TOK / 8, 256, 0, stream>>>(x, ln1_w, ln1_b, bufA, 1);
    // QKV GEMM
    k_gemm<EP_BF16, CDIM><<<dim3(QKVN / 128, TOK / 128), 128, 0, stream>>>(
        bufA, wq, qkv_b, QKVN, bufQKV, nullptr, nullptr);
    // windowed attention
    k_attn<<<NWIN * 8, 32, 0, stream>>>(bufQKV, comb, bufO);
    // proj GEMM + window reverse + roll(+3,+3) + residual -> x1
    k_gemm<EP_PROJ, CDIM><<<dim3(CDIM / 128, TOK / 128), 128, 0, stream>>>(
        bufO, wp, proj_b, CDIM, nullptr, x1, x);
    // MLP: LN2 -> fc1+GELU -> LN3 (in place) -> fc2 + residual
    k_ln256<<<TOK / 8, 256, 0, stream>>>(x1, ln2_w, ln2_b, bufA, 0);
    k_gemm<EP_GELU, CDIM><<<dim3(FFN / 128, TOK / 128), 128, 0, stream>>>(
        bufA, wf1, fc1_b, FFN, bufQKV, nullptr, nullptr);
    k_ln1024<<<TOK / 8, 256, 0, stream>>>(bufQKV, ln3_w, ln3_b);
    k_gemm<EP_FC2, FFN><<<dim3(CDIM / 128, TOK / 128), 128, 0, stream>>>(
        bufQKV, wf2, fc2_b, CDIM, nullptr, (float*)d_out, x1);
}
